// Appropriateness_Discriminator_88064009437970
// MI455X (gfx1250) — matrix operand order repaired and checked by reading, not compiled, not run on hardware
//
#include <hip/hip_runtime.h>

typedef __attribute__((ext_vector_type(16))) _Float16 v16h;
typedef __attribute__((ext_vector_type(8)))  float    v8f;
typedef __attribute__((ext_vector_type(2)))  _Float16 h2;

#define D_      128
#define T_      512
#define B_      64
#define KP_     544          // keys padded: 1 (pf) + 512 + 31 pad -> 17*32
#define NVALID_ 513
#define ROWS_   (B_ * T_)    // 32768 flat rows
#define TOT_    ROWS_
#define WSTR_   132          // padded LDS weight row stride (halves), dodges bank conflicts

// ---------------------------------------------------------------- WMMA helpers

__device__ __forceinline__ v8f wmma_f16(v16h a, v16h b, v8f c) {
  return __builtin_amdgcn_wmma_f32_16x16x32_f16(false, a, false, b, (short)0, c, false, false);
}

__device__ __forceinline__ v8f zero8() {
  v8f z;
#pragma unroll
  for (int i = 0; i < 8; ++i) z[i] = 0.f;
  return z;
}

// A-fragment 16x32 f16, row-major source with leading dim ld (halves).
// lane L: M = L%16, g = 8*(L/16); halves 0..7 = A[M][g+i], 8..15 = A[M][g+16+i]
__device__ __forceinline__ v16h load_a_frag(const _Float16* A, int ld, int lane) {
  int M = lane & 15, g = (lane >> 4) * 8;
  const _Float16* row = A + (size_t)M * ld;
  v16h a;
#pragma unroll
  for (int i = 0; i < 8; ++i) a[i] = row[g + i];
#pragma unroll
  for (int i = 0; i < 8; ++i) a[8 + i] = row[g + 16 + i];
  return a;
}

// B-fragment 32x16 where logical B[k][n] = Src[n][k] (Src row-major, ld in halves).
// lane L: N = L%16, base = 8*(L/16); halves 0..7 = Src[N][base+i], 8..15 = Src[N][base+16+i]
__device__ __forceinline__ v16h load_bT_frag(const _Float16* Src, int ld, int lane) {
  int N = lane & 15, base = (lane >> 4) * 8;
  const _Float16* p = Src + (size_t)N * ld + base;
  v16h b;
#pragma unroll
  for (int i = 0; i < 16; ++i) b[i] = p[i < 8 ? i : i + 8];
  return b;
}

__device__ __forceinline__ float redmax16(float v) {
#pragma unroll
  for (int off = 1; off < 16; off <<= 1) v = fmaxf(v, __shfl_xor(v, off, 16));
  return v;
}
__device__ __forceinline__ float redsum16(float v) {
#pragma unroll
  for (int off = 1; off < 16; off <<= 1) v += __shfl_xor(v, off, 16);
  return v;
}
__device__ __forceinline__ float sigm(float x) { return 1.f / (1.f + __expf(-x)); }

// ---------------------------------------------------------------- small kernels

__global__ __launch_bounds__(256) void k_f32_to_f16(const float* __restrict__ s,
                                                    _Float16* __restrict__ d, int n) {
  int i = blockIdx.x * 256 + threadIdx.x;
  if (i < n) d[i] = (_Float16)s[i];
}

__global__ void k_init_flags(int* flags) {
  if (threadIdx.x < 3) flags[threadIdx.x] = 0;
}

// dst[r][j] = bias[j] + sum_k W[j][k] * src[r][k]   (one block per row, 128 threads)
__global__ __launch_bounds__(128) void k_embed_rows(const float* __restrict__ src,
                                                    const float* __restrict__ W,
                                                    const float* __restrict__ bias,
                                                    int in_d, _Float16* __restrict__ dst) {
  __shared__ float xs[64];
  int r = blockIdx.x, j = threadIdx.x;
  if (j < in_d) xs[j] = src[(size_t)r * in_d + j];
  __syncthreads();
  float acc = bias[j];
  for (int k = 0; k < in_d; ++k) acc += W[j * in_d + k] * xs[k];
  dst[(size_t)r * D_ + j] = (_Float16)acc;
}

// speaker branch: row 0 = 1e-5*pf, rows 1..512 = embed(spk[b/4][t]), rows 513.. = 0
__global__ __launch_bounds__(128) void k_embed_spk(const float* __restrict__ spk,
                                                   const float* __restrict__ pf,
                                                   const float* __restrict__ W,
                                                   const float* __restrict__ bias,
                                                   int in_d, _Float16* __restrict__ dst) {
  __shared__ float xs[64];
  int idx = blockIdx.x, b = idx / KP_, kp = idx % KP_, j = threadIdx.x;
  float out;
  if (kp == 0) {
    out = 1e-5f * pf[(b >> 2) * D_ + j];
  } else if (kp <= 512) {
    if (j < in_d) xs[j] = spk[((size_t)(b >> 2) * T_ + (kp - 1)) * in_d + j];
    __syncthreads();
    out = bias[j];
    for (int k = 0; k < in_d; ++k) out += W[j * in_d + k] * xs[k];
  } else {
    out = 0.f;
  }
  dst[((size_t)b * KP_ + kp) * D_ + j] = (_Float16)out;
}

// Y[rows x 128] = X[rows x 128] @ W^T + bias  (f16 in/out, f32 accumulate, WMMA)
__global__ __launch_bounds__(32) void k_linear128(const _Float16* __restrict__ X,
                                                  const _Float16* __restrict__ W16,
                                                  const float* __restrict__ bias,
                                                  _Float16* __restrict__ Y) {
  int lane = threadIdx.x;
  int m0 = blockIdx.x * 16;
  int N = lane & 15, grp = lane >> 4;
  v16h a[4];
#pragma unroll
  for (int kc = 0; kc < 4; ++kc) a[kc] = load_a_frag(X + (size_t)m0 * D_ + kc * 32, D_, lane);
  for (int nt = 0; nt < 8; ++nt) {
    v8f acc = zero8();
#pragma unroll
    for (int kc = 0; kc < 4; ++kc)
      acc = wmma_f16(a[kc], load_bT_frag(W16 + (size_t)(nt * 16) * D_ + kc * 32, D_, lane), acc);
    float bv = bias[nt * 16 + N];
#pragma unroll
    for (int i = 0; i < 8; ++i)
      Y[(size_t)(m0 + grp * 8 + i) * D_ + nt * 16 + N] = (_Float16)(acc[i] + bv);
  }
}

// Same as k_linear128 but stores Y TRANSPOSED per batch: Yt[b][feat][row], ld = KP_.
// Gives attention's PV B-fragments fully contiguous (b128-vectorizable) loads.
__global__ __launch_bounds__(32) void k_linear128t(const _Float16* __restrict__ X,
                                                   const _Float16* __restrict__ W16,
                                                   const float* __restrict__ bias,
                                                   _Float16* __restrict__ Yt) {
  int lane = threadIdx.x;
  int m0 = blockIdx.x * 16;
  int N = lane & 15, grp = lane >> 4;
  int b = m0 / KP_;
  int rbase = m0 - b * KP_ + grp * 8;
  v16h a[4];
#pragma unroll
  for (int kc = 0; kc < 4; ++kc) a[kc] = load_a_frag(X + (size_t)m0 * D_ + kc * 32, D_, lane);
  for (int nt = 0; nt < 8; ++nt) {
    v8f acc = zero8();
#pragma unroll
    for (int kc = 0; kc < 4; ++kc)
      acc = wmma_f16(a[kc], load_bT_frag(W16 + (size_t)(nt * 16) * D_ + kc * 32, D_, lane), acc);
    float bv = bias[nt * 16 + N];
    _Float16* dst = Yt + (size_t)b * D_ * KP_ + (size_t)(nt * 16 + N) * KP_ + rbase;
#pragma unroll
    for (int i = 0; i < 8; ++i) dst[i] = (_Float16)(acc[i] + bv);
  }
}

// ---------------------------------------------------------------- flash attention
// one wave per (batch, 16-query tile); K row-major, V transposed; mask at >= NVALID_
__global__ __launch_bounds__(32) void k_attention(const _Float16* __restrict__ Q,
                                                  const _Float16* __restrict__ K,
                                                  const _Float16* __restrict__ Vt,
                                                  _Float16* __restrict__ O) {
  __shared__ _Float16 Plds[16 * 33];
  const float SCALE = 0.0883883476f;  // 1/sqrt(128)
  int lane = threadIdx.x;
  int b = blockIdx.x >> 5;        // 32 q-tiles per batch
  int qt = blockIdx.x & 31;
  int N = lane & 15, grp = lane >> 4;
  const _Float16* Qb = Q + ((size_t)b * T_ + qt * 16) * D_;
  const _Float16* Kb = K + (size_t)b * KP_ * D_;
  const _Float16* Vtb = Vt + (size_t)b * D_ * KP_;

  v16h aq[4];
#pragma unroll
  for (int kc = 0; kc < 4; ++kc) aq[kc] = load_a_frag(Qb + kc * 32, D_, lane);

  v8f o[8];
#pragma unroll
  for (int f = 0; f < 8; ++f) o[f] = zero8();
  float m8[8], l8[8], corr8[8];
#pragma unroll
  for (int i = 0; i < 8; ++i) { m8[i] = -1e30f; l8[i] = 0.f; }

  for (int c = 0; c < 17; ++c) {
    int key0 = c * 32;
    v8f s0 = zero8(), s1 = zero8();
#pragma unroll
    for (int kc = 0; kc < 4; ++kc) {
      s0 = wmma_f16(aq[kc], load_bT_frag(Kb + (size_t)key0 * D_ + kc * 32, D_, lane), s0);
      s1 = wmma_f16(aq[kc], load_bT_frag(Kb + (size_t)(key0 + 16) * D_ + kc * 32, D_, lane), s1);
    }
    int k0i = key0 + N, k1i = key0 + 16 + N;
#pragma unroll
    for (int i = 0; i < 8; ++i) {
      float v0 = (k0i < NVALID_) ? s0[i] * SCALE : -1e30f;
      float v1 = (k1i < NVALID_) ? s1[i] * SCALE : -1e30f;
      float mx = redmax16(fmaxf(v0, v1));
      float nm = fmaxf(m8[i], mx);
      float corr = __expf(m8[i] - nm);
      float p0 = __expf(v0 - nm), p1 = __expf(v1 - nm);
      float rs = redsum16(p0 + p1);
      l8[i] = l8[i] * corr + rs;
      m8[i] = nm;
      corr8[i] = corr;
      int m = grp * 8 + i;
      Plds[m * 33 + N] = (_Float16)p0;
      Plds[m * 33 + 16 + N] = (_Float16)p1;
    }
    __syncthreads();
    asm volatile("s_wait_dscnt 0" ::: "memory");
    v16h ap = load_a_frag(Plds, 33, lane);
#pragma unroll
    for (int fc = 0; fc < 8; ++fc) {
#pragma unroll
      for (int i = 0; i < 8; ++i) o[fc][i] *= corr8[i];
      v16h bv = load_bT_frag(Vtb + (size_t)(fc * 16) * KP_ + key0, KP_, lane);
      o[fc] = wmma_f16(ap, bv, o[fc]);
    }
    __syncthreads();
  }
#pragma unroll
  for (int fc = 0; fc < 8; ++fc)
#pragma unroll
    for (int i = 0; i < 8; ++i) {
      float val = o[fc][i] / l8[i];
      O[((size_t)b * T_ + qt * 16 + grp * 8 + i) * D_ + fc * 16 + N] = (_Float16)val;
    }
}

// enc[t*64+b] = concat(emo,dmm)[b*512+t] @ W_fus^T + b_fus   (writes in flat/scan order)
__global__ __launch_bounds__(32) void k_fusion(const _Float16* __restrict__ emo,
                                               const _Float16* __restrict__ dmm,
                                               const _Float16* __restrict__ Wfus16,
                                               const float* __restrict__ bfus,
                                               _Float16* __restrict__ enc) {
  int lane = threadIdx.x;
  int m0 = blockIdx.x * 16;
  int N = lane & 15, grp = lane >> 4;
  int b = m0 / T_;
  v16h a[8];
#pragma unroll
  for (int c = 0; c < 4; ++c) a[c] = load_a_frag(emo + (size_t)m0 * D_ + c * 32, D_, lane);
#pragma unroll
  for (int c = 0; c < 4; ++c) a[4 + c] = load_a_frag(dmm + (size_t)m0 * D_ + c * 32, D_, lane);
  for (int nt = 0; nt < 8; ++nt) {
    v8f acc = zero8();
#pragma unroll
    for (int c = 0; c < 8; ++c)
      acc = wmma_f16(a[c], load_bT_frag(Wfus16 + (size_t)(nt * 16) * 256 + c * 32, 256, lane), acc);
    float bv = bfus[nt * 16 + N];
#pragma unroll
    for (int i = 0; i < 8; ++i) {
      int r = m0 + grp * 8 + i;
      int t = r & (T_ - 1);
      enc[((size_t)t * B_ + b) * D_ + nt * 16 + N] = (_Float16)(acc[i] + bv);
    }
  }
}

// Xp0[32768 x 512] = enc @ W_ih0^T + b_ih0   (pre-computes layer-0 input transform)
__global__ __launch_bounds__(32) void k_xproj(const _Float16* __restrict__ enc,
                                              const _Float16* __restrict__ Wih0,
                                              const float* __restrict__ b_ih,
                                              float* __restrict__ Xp0) {
  int lane = threadIdx.x;
  int m0 = blockIdx.x * 16;
  int N = lane & 15, grp = lane >> 4;
  v16h a[4];
#pragma unroll
  for (int kc = 0; kc < 4; ++kc) a[kc] = load_a_frag(enc + (size_t)m0 * D_ + kc * 32, D_, lane);
  for (int nt = 0; nt < 32; ++nt) {
    v8f acc = zero8();
#pragma unroll
    for (int kc = 0; kc < 4; ++kc)
      acc = wmma_f16(a[kc], load_bT_frag(Wih0 + (size_t)(nt * 16) * D_ + kc * 32, D_, lane), acc);
    float bv = b_ih[nt * 16 + N];
#pragma unroll
    for (int i = 0; i < 8; ++i)
      Xp0[(size_t)(m0 + grp * 8 + i) * 512 + nt * 16 + N] = acc[i] + bv;
  }
}

// ---------------------------------------------------------------- pipelined LSTM
// 3 persistent workgroups (one layer each). f16-pair weights+activations in LDS
// (v_fma_mix dots, f32 accumulate). Recurrent dot overlaps the upstream spin-wait.
__global__ __launch_bounds__(1024) void k_lstm(const float* __restrict__ W_ih,
                                               const float* __restrict__ W_hh,
                                               const float* __restrict__ b_ih,
                                               const float* __restrict__ b_hh,
                                               const float* __restrict__ Xp0,
                                               float* __restrict__ Hbuf,   // 2 * TOT_ * 128
                                               float* __restrict__ lastH,  // 64 * 128
                                               int* __restrict__ flags) {
  extern __shared__ char smem[];
  _Float16* Wih = (_Float16*)smem;                 // 512 * WSTR_
  _Float16* Whh = Wih + 512 * WSTR_;               // 512 * WSTR_
  float* gA = (float*)(Whh + 512 * WSTR_);         // 512
  float* gB = gA + 512;                            // 512
  float* cv = gB + 512;                            // 128
  _Float16* xh16 = (_Float16*)(cv + 128);          // 128 (f16 pairs)
  _Float16* hh16 = xh16 + 128;                     // 128 (f16 pairs)

  int l = blockIdx.x, tid = threadIdx.x;
  const float* Wih_g = W_ih + (size_t)l * 512 * 128;
  const float* Whh_g = W_hh + (size_t)l * 512 * 128;
  for (int idx = tid; idx < 512 * 128; idx += 1024) {
    int row = idx >> 7, col = idx & 127;
    Wih[row * WSTR_ + col] = (_Float16)Wih_g[idx];
    Whh[row * WSTR_ + col] = (_Float16)Whh_g[idx];
  }
  if (tid < 128) { cv[tid] = 0.f; hh16[tid] = (_Float16)0.f; xh16[tid] = (_Float16)0.f; }
  const float* bih = b_ih + l * 512;
  const float* bhh = b_hh + l * 512;
  __syncthreads();

  for (int t = 0; t < TOT_; ++t) {
    // ---- phase 1: recurrent dot (W_hh . h) runs concurrently with the
    //      upstream flag spin (l>0) / Xp0 stream load + prefetch (l==0)
    if (tid >= 512) {
      int j = tid - 512;
      float acc = bhh[j];
      const h2* wr = (const h2*)(Whh + j * WSTR_);
      const h2* hp = (const h2*)hh16;
#pragma unroll 8
      for (int k = 0; k < 64; ++k) {
        h2 w = wr[k], x = hp[k];
        acc += (float)w[0] * (float)x[0] + (float)w[1] * (float)x[1];
      }
      gB[j] = acc;
    } else if (l == 0) {
      if (t + 16 < TOT_) __builtin_prefetch(Xp0 + (size_t)(t + 16) * 512 + tid, 0, 1);
      gA[tid] = Xp0[(size_t)t * 512 + tid];
    } else if (tid == 0) {
      while (__hip_atomic_load(&flags[l - 1], __ATOMIC_ACQUIRE, __HIP_MEMORY_SCOPE_AGENT) <= t)
        __builtin_amdgcn_s_sleep(1);
    }
    __syncthreads();
    // ---- phase 2 (l>0 only): pull x = h_{l-1}(t), then input dot (W_ih . x)
    if (l > 0) {
      if (tid < 128)
        xh16[tid] = (_Float16)Hbuf[(size_t)(l - 1) * TOT_ * 128 + (size_t)t * 128 + tid];
      __syncthreads();
      if (tid < 512) {
        int j = tid;
        float acc = bih[j];
        const h2* wr = (const h2*)(Wih + j * WSTR_);
        const h2* xp = (const h2*)xh16;
#pragma unroll 8
        for (int k = 0; k < 64; ++k) {
          h2 w = wr[k], x = xp[k];
          acc += (float)w[0] * (float)x[0] + (float)w[1] * (float)x[1];
        }
        gA[j] = acc;
      }
      __syncthreads();
    }
    // ---- activation
    if (tid < 128) {
      int n = tid;
      float ig = gA[n] + gB[n];
      float fg = gA[128 + n] + gB[128 + n];
      float gg = gA[256 + n] + gB[256 + n];
      float og = gA[384 + n] + gB[384 + n];
      float cc = sigm(fg) * cv[n] + sigm(ig) * tanhf(gg);
      float hh = sigm(og) * tanhf(cc);
      cv[n] = cc;
      hh16[n] = (_Float16)hh;
      if (l < 2) {
        Hbuf[(size_t)l * TOT_ * 128 + (size_t)t * 128 + n] = hh;
        __threadfence();
      } else if (t >= TOT_ - 64) {
        lastH[(t - (TOT_ - 64)) * 128 + n] = hh;
      }
    }
    __syncthreads();
    if (l < 2 && tid == 0)
      __hip_atomic_store(&flags[l], t + 1, __ATOMIC_RELEASE, __HIP_MEMORY_SCOPE_AGENT);
  }
}

// out[r] = sigmoid( relu(last[r] @ W1^T + b1) @ W2^T + b2 )
__global__ __launch_bounds__(128) void k_fc(const float* __restrict__ lastH,
                                            const float* __restrict__ W1,
                                            const float* __restrict__ b1,
                                            const float* __restrict__ W2,
                                            const float* __restrict__ b2,
                                            float* __restrict__ out) {
  __shared__ float xr[128];
  __shared__ float red[128];
  int r = blockIdx.x, j = threadIdx.x;
  xr[j] = lastH[r * 128 + j];
  __syncthreads();
  float acc = b1[j];
  for (int k = 0; k < 128; ++k) acc += W1[j * 128 + k] * xr[k];
  acc = fmaxf(acc, 0.f);
  red[j] = acc * W2[j];
  __syncthreads();
  for (int s = 64; s > 0; s >>= 1) {
    if (j < s) red[j] += red[j + s];
    __syncthreads();
  }
  if (j == 0) out[r] = 1.f / (1.f + expf(-(red[0] + b2[0])));
}

// ---------------------------------------------------------------- host launcher

extern "C" void kernel_launch(void* const* d_in, const int* in_sizes, int n_in,
                              void* d_out, int out_size, void* d_ws, size_t ws_size,
                              hipStream_t stream) {
  (void)in_sizes; (void)n_in; (void)out_size; (void)ws_size;
  const float* sp_emo = (const float*)d_in[0];
  const float* li_emo = (const float*)d_in[1];
  const float* sp_dmm = (const float*)d_in[2];
  const float* li_dmm = (const float*)d_in[3];
  const float* pf     = (const float*)d_in[4];
  const float* W_em  = (const float*)d_in[6];  const float* b_em  = (const float*)d_in[7];
  const float* W_3d  = (const float*)d_in[8];  const float* b_3d  = (const float*)d_in[9];
  const float* Wq_e  = (const float*)d_in[10]; const float* bq_e  = (const float*)d_in[11];
  const float* Wk_e  = (const float*)d_in[12]; const float* bk_e  = (const float*)d_in[13];
  const float* Wv_e  = (const float*)d_in[14]; const float* bv_e  = (const float*)d_in[15];
  const float* Wq_d  = (const float*)d_in[16]; const float* bq_d  = (const float*)d_in[17];
  const float* Wk_d  = (const float*)d_in[18]; const float* bk_d  = (const float*)d_in[19];
  const float* Wv_d  = (const float*)d_in[20]; const float* bv_d  = (const float*)d_in[21];
  const float* W_fus = (const float*)d_in[22]; const float* b_fus = (const float*)d_in[23];
  const float* W_fc1 = (const float*)d_in[24]; const float* b_fc1 = (const float*)d_in[25];
  const float* W_fc2 = (const float*)d_in[26]; const float* b_fc2 = (const float*)d_in[27];
  const float* W_ih  = (const float*)d_in[28]; const float* W_hh  = (const float*)d_in[29];
  const float* b_ih  = (const float*)d_in[30]; const float* b_hh  = (const float*)d_in[31];
  float* out = (float*)d_out;
  char* ws = (char*)d_ws;

  size_t off = 0;
  auto bump = [&](size_t bytes) { size_t o = off; off += (bytes + 255) & ~(size_t)255; return o; };
  // persistent buffers
  size_t o_emo16  = bump((size_t)ROWS_ * D_ * 2);
  size_t o_dmm16  = bump((size_t)ROWS_ * D_ * 2);
  size_t o_enc16  = bump((size_t)ROWS_ * D_ * 2);
  size_t o_wqe16  = bump(16384 * 2);
  size_t o_wke16  = bump(16384 * 2);
  size_t o_wve16  = bump(16384 * 2);
  size_t o_wqd16  = bump(16384 * 2);
  size_t o_wkd16  = bump(16384 * 2);
  size_t o_wvd16  = bump(16384 * 2);
  size_t o_wfus16 = bump(32768 * 2);
  size_t o_wih016 = bump(65536 * 2);
  size_t o_hbuf   = bump(2ull * TOT_ * 128 * 4);
  size_t o_last   = bump(64 * 128 * 4);
  size_t o_flags  = bump(256);
  // region A: early buffers, later aliased by Xp0 (stream order makes this safe)
  size_t regionA = off;
  size_t a = regionA;
  auto bumpA = [&](size_t bytes) { size_t o = a; a += (bytes + 255) & ~(size_t)255; return o; };
  size_t o_le  = bumpA((size_t)ROWS_ * D_ * 2);
  size_t o_ld  = bumpA((size_t)ROWS_ * D_ * 2);
  size_t o_seK = bumpA((size_t)B_ * KP_ * D_ * 2);
  size_t o_sdK = bumpA((size_t)B_ * KP_ * D_ * 2);
  size_t o_qe  = bumpA((size_t)ROWS_ * D_ * 2);
  size_t o_qd  = bumpA((size_t)ROWS_ * D_ * 2);
  size_t o_ke  = bumpA((size_t)B_ * KP_ * D_ * 2);
  size_t o_ve  = bumpA((size_t)B_ * KP_ * D_ * 2);   // V-transposed (same footprint)
  size_t o_kd  = bumpA((size_t)B_ * KP_ * D_ * 2);
  size_t o_vd  = bumpA((size_t)B_ * KP_ * D_ * 2);   // V-transposed (same footprint)
  size_t o_xp0 = regionA;  // 32768*512*4 = 64 MB, fits inside region A (~87 MB)

#define WSF16(o) ((_Float16*)(ws + (o)))
  int* flags = (int*)(ws + o_flags);
  float* Xp0 = (float*)(ws + o_xp0);
  float* Hbuf = (float*)(ws + o_hbuf);
  float* lastH = (float*)(ws + o_last);

  k_init_flags<<<1, 32, 0, stream>>>(flags);

  // weight f32 -> f16 conversions
  k_f32_to_f16<<<64, 256, 0, stream>>>(Wq_e, WSF16(o_wqe16), 16384);
  k_f32_to_f16<<<64, 256, 0, stream>>>(Wk_e, WSF16(o_wke16), 16384);
  k_f32_to_f16<<<64, 256, 0, stream>>>(Wv_e, WSF16(o_wve16), 16384);
  k_f32_to_f16<<<64, 256, 0, stream>>>(Wq_d, WSF16(o_wqd16), 16384);
  k_f32_to_f16<<<64, 256, 0, stream>>>(Wk_d, WSF16(o_wkd16), 16384);
  k_f32_to_f16<<<64, 256, 0, stream>>>(Wv_d, WSF16(o_wvd16), 16384);
  k_f32_to_f16<<<128, 256, 0, stream>>>(W_fus, WSF16(o_wfus16), 32768);
  k_f32_to_f16<<<256, 256, 0, stream>>>(W_ih, WSF16(o_wih016), 65536);  // layer 0 slab

  // embeddings
  k_embed_rows<<<ROWS_, 128, 0, stream>>>(li_emo, W_em, b_em, 25, WSF16(o_le));
  k_embed_rows<<<ROWS_, 128, 0, stream>>>(li_dmm, W_3d, b_3d, 58, WSF16(o_ld));
  k_embed_spk<<<B_ * KP_, 128, 0, stream>>>(sp_emo, pf, W_em, b_em, 25, WSF16(o_seK));
  k_embed_spk<<<B_ * KP_, 128, 0, stream>>>(sp_dmm, pf, W_3d, b_3d, 58, WSF16(o_sdK));

  // QKV projections (WMMA); V stored transposed for attention PV coalescing
  k_linear128<<<ROWS_ / 16, 32, 0, stream>>>(WSF16(o_le), WSF16(o_wqe16), bq_e, WSF16(o_qe));
  k_linear128<<<B_ * KP_ / 16, 32, 0, stream>>>(WSF16(o_seK), WSF16(o_wke16), bk_e, WSF16(o_ke));
  k_linear128t<<<B_ * KP_ / 16, 32, 0, stream>>>(WSF16(o_seK), WSF16(o_wve16), bv_e, WSF16(o_ve));
  k_linear128<<<ROWS_ / 16, 32, 0, stream>>>(WSF16(o_ld), WSF16(o_wqd16), bq_d, WSF16(o_qd));
  k_linear128<<<B_ * KP_ / 16, 32, 0, stream>>>(WSF16(o_sdK), WSF16(o_wkd16), bk_d, WSF16(o_kd));
  k_linear128t<<<B_ * KP_ / 16, 32, 0, stream>>>(WSF16(o_sdK), WSF16(o_wvd16), bv_d, WSF16(o_vd));

  // flash attention (WMMA)
  k_attention<<<B_ * 32, 32, 0, stream>>>(WSF16(o_qe), WSF16(o_ke), WSF16(o_ve), WSF16(o_emo16));
  k_attention<<<B_ * 32, 32, 0, stream>>>(WSF16(o_qd), WSF16(o_kd), WSF16(o_vd), WSF16(o_dmm16));

  // fusion + layer-0 input projection (WMMA)
  k_fusion<<<ROWS_ / 16, 32, 0, stream>>>(WSF16(o_emo16), WSF16(o_dmm16), WSF16(o_wfus16), b_fus,
                                          WSF16(o_enc16));
  k_xproj<<<ROWS_ / 16, 32, 0, stream>>>(WSF16(o_enc16), WSF16(o_wih016), b_ih, Xp0);

  // pipelined 3-layer sequential LSTM (3 WGPs, LDS-resident f16 weights)
  size_t lstm_smem = 2ull * 512 * WSTR_ * 2 + (512 + 512 + 128) * 4 + 2 * 128 * 2;
  k_lstm<<<3, 1024, lstm_smem, stream>>>(W_ih, W_hh, b_ih, b_hh, Xp0, Hbuf, lastH, flags);

  // classifier head
  k_fc<<<64, 128, 0, stream>>>(lastH, W_fc1, b_fc1, W_fc2, b_fc2, out);
#undef WSF16
}
